// TreeLSTM_5308579578292
// MI455X (gfx1250) — compile-verified, matrix operand order checked by reading, not run on hardware
//
#include <hip/hip_runtime.h>
#include <hip/hip_bf16.h>

typedef __attribute__((ext_vector_type(16))) _Float16 v16h;
typedef __attribute__((ext_vector_type(8)))  float    v8f;

#define D_FEAT   64
#define GD       256        // 4*D
#define OPS      15
#define NTREE    3
#define DEPTH    10
#define LEAVES   1024
#define INTERNAL 1023
#define MAXNODES 1536       // 3 * 512 (level 9)

// ---------- helpers ----------

__device__ __forceinline__ float sigm(float x) { return 1.0f / (1.0f + __expf(-x)); }

// A-operand LDS swizzle: value (row r, concat-k k in [0,128)) -> half offset in sSx
// Layout matches V_WMMA 16-bit A operand: lane L holds row L&15,
// kbase = (L<16)?0:8, half j -> K = kbase + (j<8 ? j : 8+j) within a 32-wide K chunk.
__device__ __forceinline__ int sxOff(int r, int k) {
    int kc  = k >> 5;
    int kk  = k & 31;
    int grp = (kk >> 3) & 1;                       // lanes 16..31 hold kbase=8 halves
    int j   = (kk & 7) | ((kk >> 4) << 3);
    return ((kc << 5) + (grp << 4) + r) * 16 + j;  // (chunk,lane)*16 + j
}

// ---------- setup kernels ----------

// wB layout (halves): ((((layer*15+op)*16 + nt)*4 + kc)*32 + lane)*16 + j
// element: col n = lane&15 -> gate g = nt*16+n ; kbase=(lane<16)?0:8 ;
//          kk = (j<8)? kbase+j : kbase+8+j ; k = kc*32+kk ; k<64 -> w_ih else w_hh
__global__ void convert_weights(const float* __restrict__ w_ih,
                                const float* __restrict__ w_hh,
                                _Float16* __restrict__ wB) {
    int tid = blockIdx.x * 256 + threadIdx.x;      // < 983040
    int j     = tid & 15;
    int lane  = (tid >> 4) & 31;
    int kc    = (tid >> 9) & 3;
    int nt    = (tid >> 11) & 15;
    int lo    = tid >> 15;                          // 0..29 = layer*15+op
    int ncol  = lane & 15;
    int g     = nt * 16 + ncol;
    int kbase = (lane < 16) ? 0 : 8;
    int kk    = (j < 8) ? (kbase + j) : (kbase + 8 + j);
    int k     = kc * 32 + kk;
    size_t wi = ((size_t)lo * GD + g) * D_FEAT;
    float v = (k < D_FEAT) ? w_ih[wi + k] : w_hh[wi + (k - D_FEAT)];
    wB[tid] = (_Float16)v;
}

__global__ void combine_bias(const float* __restrict__ b_ih,
                             const float* __restrict__ b_hh,
                             float* __restrict__ bias) {
    int tid = blockIdx.x * 256 + threadIdx.x;      // < 7680
    if (tid < 2 * OPS * GD) bias[tid] = b_ih[tid] + b_hh[tid];
}

__global__ void zero_counts(int* cnt) {
    int t = threadIdx.x;
    if (t < DEPTH * OPS) cnt[t] = 0;
}

// group internal nodes by (level, op); node id within level = tree*n + i
__global__ void build_groups(const int* __restrict__ op_ids,
                             int* __restrict__ cnt,
                             int* __restrict__ idx) {
    int tid = blockIdx.x * 256 + threadIdx.x;
    if (tid >= NTREE * INTERNAL) return;
    int tree  = tid / INTERNAL;
    int jj    = tid - tree * INTERNAL;
    int level = 31 - __builtin_clz(jj + 1);
    int base  = (1 << level) - 1;
    int i     = jj - base;
    int m     = tree * (1 << level) + i;
    int op    = op_ids[tree * INTERNAL + jj];
    int slot  = atomicAdd(&cnt[level * OPS + op], 1);
    idx[(level * OPS + op) * MAXNODES + slot] = m;
}

__global__ void gather_leaves(const float* __restrict__ sym,
                              const int* __restrict__ leaf_ids,
                              float* __restrict__ bufA) {
    int tid = blockIdx.x * 256 + threadIdx.x;      // < 3*1024*64
    int d   = tid & 63;
    int row = tid >> 6;                             // tree*1024 + leaf
    int id  = leaf_ids[row];
    bufA[tid] = sym[(size_t)id * D_FEAT + d];
}

// ---------- level kernel: 16 same-op nodes per block, 4 waves ----------

__global__ __launch_bounds__(128) void level_kernel(
    int n,                                // nodes per tree at this level
    const float* __restrict__ child,      // [3*2n][64]
    float* __restrict__ outBuf,           // [3*n][64]
    const _Float16* __restrict__ wB,
    const float* __restrict__ bias,
    const int* __restrict__ cnt,          // + level*15 (host offset)
    const int* __restrict__ idx)          // + level*15*MAXNODES (host offset)
{
    int op   = blockIdx.x;
    int tile = blockIdx.y;
    int count = cnt[op];
    if (tile * 16 >= count) return;       // uniform exit

    __shared__ int nodes[16];
    __shared__ int c0row[16];
    __shared__ int c1row[16];
    alignas(32) __shared__ _Float16 sSx[4 * 32 * 16];   // A operand, swizzled, 4KB
    __shared__ float sG[16 * GD];                       // gates, 16KB
    __shared__ float sC[16 * D_FEAT];                   // cell state, 4KB
    __shared__ float sH1[3 * 16 * D_FEAT];              // layer-1 hiddens, 12KB

    int tid  = threadIdx.x;
    int wave = tid >> 5;
    int lane = tid & 31;

    if (tid < 16) {
        int s = tile * 16 + tid;
        int m = (s < count) ? idx[op * MAXNODES + s] : -1;
        nodes[tid] = m;
        if (m >= 0) {
            int tree = m / n;
            int i    = m - tree * n;
            c0row[tid] = tree * (2 * n) + 2 * i;
            c1row[tid] = c0row[tid] + 1;
        } else {
            c0row[tid] = -1; c1row[tid] = -1;
        }
    }
    __syncthreads();

    for (int layer = 0; layer < 2; ++layer) {
        const _Float16* wbase = wB + (size_t)(layer * OPS + op) * (16 * 4 * 32 * 16);
        const float*    bbase = bias + (layer * OPS + op) * GD;
        __builtin_prefetch(wbase + (size_t)tid * 256, 0, 0);   // global_prefetch_b8

        // init c = 0, h-region = 0, x-region for t=0 (ones for layer0, h1[t=0] for layer1)
        for (int e = tid; e < 16 * D_FEAT; e += 128) {
            int r = e >> 6, d = e & 63;
            sC[e] = 0.0f;
            float xv = (layer == 0) ? 1.0f : sH1[e];           // sH1[t=0]
            sSx[sxOff(r, d)]          = (_Float16)xv;
            sSx[sxOff(r, 64 + d)]     = (_Float16)0.0f;
        }
        __syncthreads();

        for (int t = 0; t < 3; ++t) {
            // ---- GEMM: G[16,256] = S[16,128] @ Wcat^T, K in 4 chunks of 32 ----
            v16h a0 = *(const v16h*)(sSx + (0 * 32 + lane) * 16);
            v16h a1 = *(const v16h*)(sSx + (1 * 32 + lane) * 16);
            v16h a2 = *(const v16h*)(sSx + (2 * 32 + lane) * 16);
            v16h a3 = *(const v16h*)(sSx + (3 * 32 + lane) * 16);
            #pragma unroll
            for (int s = 0; s < 4; ++s) {
                int nt = wave * 4 + s;
                const v16h* bp = (const v16h*)(wbase + (size_t)nt * 2048) + lane;
                v8f acc = {};
                acc = __builtin_amdgcn_wmma_f32_16x16x32_f16(false, a0, false, bp[0 * 32],
                                                             (short)0, acc, false, false);
                acc = __builtin_amdgcn_wmma_f32_16x16x32_f16(false, a1, false, bp[1 * 32],
                                                             (short)0, acc, false, false);
                acc = __builtin_amdgcn_wmma_f32_16x16x32_f16(false, a2, false, bp[2 * 32],
                                                             (short)0, acc, false, false);
                acc = __builtin_amdgcn_wmma_f32_16x16x32_f16(false, a3, false, bp[3 * 32],
                                                             (short)0, acc, false, false);
                // D layout: lane col = lane&15, rows (lane>>4)*8 .. +7 in acc[0..7]
                int ncol  = lane & 15;
                int mbase = (lane >> 4) * 8;
                int g     = nt * 16 + ncol;
                float bv  = bbase[g];
                #pragma unroll
                for (int v = 0; v < 8; ++v)
                    sG[(mbase + v) * GD + g] = acc[v] + bv;
            }
            __syncthreads();

            // ---- LSTM cell update (torch gate order i,f,g,o) ----
            for (int e = tid; e < 16 * D_FEAT; e += 128) {
                int r = e >> 6, d = e & 63;
                float gi = sG[r * GD + d];
                float gf = sG[r * GD + 64 + d];
                float gg = sG[r * GD + 128 + d];
                float go = sG[r * GD + 192 + d];
                float cc = sigm(gf) * sC[e] + sigm(gi) * tanhf(gg);
                float hh = sigm(go) * tanhf(cc);
                sC[e] = cc;
                sSx[sxOff(r, 64 + d)] = (_Float16)hh;          // h feeds next step
                if (layer == 0) sH1[t * (16 * D_FEAT) + e] = hh;
                if (t < 2) {
                    float xv;
                    if (layer == 0) {
                        int crow = (t == 0) ? c0row[r] : c1row[r];
                        xv = (crow >= 0) ? child[(size_t)crow * D_FEAT + d] : 0.0f;
                    } else {
                        xv = sH1[(t + 1) * (16 * D_FEAT) + e];
                    }
                    sSx[sxOff(r, d)] = (_Float16)xv;
                } else if (layer == 1) {
                    int m = nodes[r];
                    if (m >= 0) outBuf[(size_t)m * D_FEAT + d] = hh;
                }
            }
            __syncthreads();
        }
    }
}

// ---------- final cosine/softmax/combine ----------

__global__ void final_kernel(const float* __restrict__ roots,   // [3][64]
                             const float* __restrict__ att,     // [64]
                             float* __restrict__ out) {         // [64]
    __shared__ float red[64];
    __shared__ float res[8];
    int d = threadIdx.x;                                        // 64 threads
    float a  = att[d];
    float r0 = roots[d], r1 = roots[64 + d], r2 = roots[128 + d];
    float vals[7] = { a * a, r0 * a, r1 * a, r2 * a, r0 * r0, r1 * r1, r2 * r2 };
    for (int q = 0; q < 7; ++q) {
        red[d] = vals[q];
        __syncthreads();
        for (int s = 32; s >= 1; s >>= 1) {
            if (d < s) red[d] += red[d + s];
            __syncthreads();
        }
        if (d == 0) res[q] = red[0];
        __syncthreads();
    }
    if (d == 0) {
        const float eps = 1e-8f;
        float na = fmaxf(sqrtf(res[0]), eps);
        float w[3], mx = -1e30f;
        for (int t = 0; t < 3; ++t) {
            w[t] = res[1 + t] / (fmaxf(sqrtf(res[4 + t]), eps) * na);
            mx = fmaxf(mx, w[t]);
        }
        float se = 0.0f;
        for (int t = 0; t < 3; ++t) { w[t] = __expf(w[t] - mx); se += w[t]; }
        for (int t = 0; t < 3; ++t) res[1 + t] = w[t] / se;
    }
    __syncthreads();
    out[d] = res[1] * r0 + res[2] * r1 + res[3] * r2;
}

// ---------- host ----------

extern "C" void kernel_launch(void* const* d_in, const int* in_sizes, int n_in,
                              void* d_out, int out_size, void* d_ws, size_t ws_size,
                              hipStream_t stream) {
    const float* symbol_table = (const float*)d_in[0];
    const float* w_ih         = (const float*)d_in[1];
    const float* w_hh         = (const float*)d_in[2];
    const float* b_ih         = (const float*)d_in[3];
    const float* b_hh         = (const float*)d_in[4];
    const float* attvec       = (const float*)d_in[5];
    const int*   leaf_ids     = (const int*)d_in[6];
    const int*   op_ids       = (const int*)d_in[7];
    float* out = (float*)d_out;

    // workspace carve-up (256B aligned)
    char* ws = (char*)d_ws;
    size_t off = 0;
    auto take = [&](size_t bytes) -> char* {
        char* p = ws + off;
        off = (off + bytes + 255) & ~(size_t)255;
        return p;
    };
    _Float16* wB    = (_Float16*)take((size_t)2 * OPS * 16 * 4 * 32 * 16 * sizeof(_Float16)); // 1.875 MB
    float*    biasW = (float*)take((size_t)2 * OPS * GD * sizeof(float));                     // 30 KB
    int*      cnt   = (int*)take((size_t)DEPTH * OPS * sizeof(int));
    int*      gidx  = (int*)take((size_t)DEPTH * OPS * MAXNODES * sizeof(int));               // 900 KB
    float*    bufA  = (float*)take((size_t)NTREE * LEAVES * D_FEAT * sizeof(float));          // 768 KB
    float*    bufB  = (float*)take((size_t)NTREE * LEAVES * D_FEAT * sizeof(float));          // 768 KB

    // setup (independent, cheap)
    zero_counts<<<1, 256, 0, stream>>>(cnt);
    convert_weights<<<3840, 256, 0, stream>>>(w_ih, w_hh, wB);           // 983040 halves
    combine_bias<<<30, 256, 0, stream>>>(b_ih, b_hh, biasW);
    build_groups<<<12, 256, 0, stream>>>(op_ids, cnt, gidx);
    gather_leaves<<<768, 256, 0, stream>>>(symbol_table, leaf_ids, bufA);

    // bottom-up levels, ping-pong buffers
    const float* src = bufA;
    float*       dst = bufB;
    for (int level = DEPTH - 1; level >= 0; --level) {
        int n = 1 << level;
        int M = NTREE * n;
        dim3 grid(OPS, (M + 15) / 16);
        level_kernel<<<grid, 128, 0, stream>>>(n, src, dst, wB, biasW,
                                               cnt + level * OPS,
                                               gidx + (size_t)level * OPS * MAXNODES);
        float* tmp = dst;
        dst = (float*)src;
        src = tmp;
    }
    // roots now at src (rows 0..2)
    final_kernel<<<1, 64, 0, stream>>>(src, attvec, out);
}
